// GCN_1056561955307
// MI455X (gfx1250) — matrix operand order repaired and checked by reading, not compiled, not run on hardware
//
#include <hip/hip_runtime.h>

#define N_NODES  50000
#define N_EDGES  800000
#define N_GRAPHS 128
#define D        128
#define D_OUT    3

typedef __attribute__((ext_vector_type(16))) __bf16 v16bf;
typedef __attribute__((ext_vector_type(8)))  __bf16 v8bf;
typedef __attribute__((ext_vector_type(8)))  float  v8f;

// ---------------------------------------------------------------------------
// 0) init: deg = 1.0 (self loop), counts = 0, pooled = 0
// ---------------------------------------------------------------------------
__global__ void init_kernel(float* __restrict__ deg, float* __restrict__ counts,
                            float* __restrict__ pooled) {
    int i = blockIdx.x * blockDim.x + threadIdx.x;
    if (i < N_NODES)        deg[i]    = 1.0f;
    if (i < N_GRAPHS)       counts[i] = 0.0f;
    if (i < N_GRAPHS * D)   pooled[i] = 0.0f;
}

// ---------------------------------------------------------------------------
// 1) accumulate in-degree over edge targets
// ---------------------------------------------------------------------------
__global__ void degree_kernel(const int* __restrict__ col, float* __restrict__ deg) {
    int e = blockIdx.x * blockDim.x + threadIdx.x;
    if (e < N_EDGES) atomicAdd(&deg[col[e]], 1.0f);
}

// ---------------------------------------------------------------------------
// 2) dinv = rsqrt(deg)  (deg >= 1 always); also per-graph node counts
// ---------------------------------------------------------------------------
__global__ void norm_kernel(float* __restrict__ deg, const int* __restrict__ batch,
                            float* __restrict__ counts) {
    int i = blockIdx.x * blockDim.x + threadIdx.x;
    if (i >= N_NODES) return;
    deg[i] = rsqrtf(deg[i]);              // in-place: deg becomes dinv
    atomicAdd(&counts[batch[i]], 1.0f);
}

// ---------------------------------------------------------------------------
// 3) pack 128x128 f32 weight into bf16 WMMA B-fragment layout:
//    wp[((t*4 + kb)*32 + lane)*16 + j] = W[kb*32 + (lane>>4)*8 + j + (j&8)][t*16 + (lane&15)]
//    grid.x selects which of the 3 weights; 1024 threads = one fragment each.
// ---------------------------------------------------------------------------
__global__ __launch_bounds__(1024) void packw_kernel(const float* __restrict__ W1,
                                                     const float* __restrict__ W2,
                                                     const float* __restrict__ W3,
                                                     __bf16* __restrict__ wp) {
    const float* W = (blockIdx.x == 0) ? W1 : (blockIdx.x == 1) ? W2 : W3;
    __bf16* out = wp + (size_t)blockIdx.x * (D * D);
    int tid  = threadIdx.x;          // 0..1023
    int t    = tid >> 7;             // n-tile 0..7
    int kb   = (tid >> 5) & 3;       // k-chunk 0..3
    int lane = tid & 31;
    int n    = t * 16 + (lane & 15);
    int kseg = kb * 32 + (lane >> 4) * 8;
    __bf16* dst = out + ((size_t)((t * 4 + kb) * 32 + lane)) * 16;
#pragma unroll
    for (int j = 0; j < 16; ++j)
        dst[j] = (__bf16)W[(kseg + j + (j & 8)) * D + n];
}

// ---------------------------------------------------------------------------
// 4) f32 -> bf16 feature conversion (first-layer A operand)
// ---------------------------------------------------------------------------
__global__ void cvt_kernel(const float* __restrict__ x, __bf16* __restrict__ hb) {
    int i = blockIdx.x * blockDim.x + threadIdx.x;
    if (i < N_NODES * D) hb[i] = (__bf16)x[i];
}

// ---------------------------------------------------------------------------
// 5) GEMM: hw[N,128] = hb[N,128] @ Wpacked  via v_wmma_f32_16x16x32_bf16.
//    One wave per 16-row M-tile (50000 = 16*3125 exactly); 8 waves/block.
// ---------------------------------------------------------------------------
__global__ __launch_bounds__(256) void gemm_kernel(const __bf16* __restrict__ hb,
                                                   const __bf16* __restrict__ wp,
                                                   float* __restrict__ hw) {
    int wave = threadIdx.x >> 5;
    int lane = threadIdx.x & 31;
    int tile = blockIdx.x * 8 + wave;
    if (tile >= N_NODES / 16) return;            // wave-uniform exit, EXEC stays full
    const int row0  = tile * 16;
    const int m     = lane & 15;                 // A row / D column within 16-tile
    const int khalf = lane >> 4;                 // 0 or 1

    const __bf16* arow = hb + (size_t)(row0 + m) * D;

    v8f acc[8] = {};
#pragma unroll
    for (int kb = 0; kb < 4; ++kb) {
        // A fragment: 16-bit A 16x32 layout -> two contiguous 8-bf16 runs per lane
        int kbase = kb * 32 + khalf * 8;
        v8bf a0 = *(const v8bf*)(arow + kbase);
        v8bf a1 = *(const v8bf*)(arow + kbase + 16);
        v16bf a;
#pragma unroll
        for (int j = 0; j < 8; ++j) { a[j] = a0[j]; a[j + 8] = a1[j]; }
#pragma unroll
        for (int t = 0; t < 8; ++t) {
            v16bf b = *(const v16bf*)(wp + ((size_t)((t * 4 + kb) * 32 + lane)) * 16);
            acc[t] = __builtin_amdgcn_wmma_f32_16x16x32_bf16(
                         false, a, false, b, (short)0, acc[t], false, false);
        }
    }
    // C/D layout: VGPR r -> M = r + khalf*8, N = lane&15
#pragma unroll
    for (int t = 0; t < 8; ++t) {
        int col = t * 16 + m;
#pragma unroll
        for (int r = 0; r < 8; ++r)
            hw[(size_t)(row0 + khalf * 8 + r) * D + col] = acc[t][r];
    }
}

// ---------------------------------------------------------------------------
// 6) zero the aggregation buffer (float4 stores)
// ---------------------------------------------------------------------------
__global__ void zero_kernel(float4* __restrict__ agg) {
    int i = blockIdx.x * blockDim.x + threadIdx.x;
    if (i < (N_NODES * D) / 4) agg[i] = make_float4(0.f, 0.f, 0.f, 0.f);
}

// ---------------------------------------------------------------------------
// 7) scatter: agg[c] += hw[r] * dinv[r]*dinv[c]   (one wave per message,
//    messages E..E+N-1 are self loops)
// ---------------------------------------------------------------------------
__global__ __launch_bounds__(256) void scatter_kernel(const float* __restrict__ hw,
                                                      float* __restrict__ agg,
                                                      const int* __restrict__ row,
                                                      const int* __restrict__ col,
                                                      const float* __restrict__ dinv) {
    int msg  = blockIdx.x * 8 + (threadIdx.x >> 5);
    int lane = threadIdx.x & 31;
    if (msg >= N_EDGES + N_NODES) return;        // wave-uniform
    int r, c;
    if (msg < N_EDGES) { r = row[msg]; c = col[msg]; }
    else               { r = c = msg - N_EDGES; }
    float nrm = dinv[r] * dinv[c];
    const float* src = hw  + (size_t)r * D;
    float*       dst = agg + (size_t)c * D;
#pragma unroll
    for (int j = 0; j < 4; ++j) {
        int d = lane + j * 32;
        atomicAdd(dst + d, src[d] * nrm);
    }
}

// ---------------------------------------------------------------------------
// 8) bias + (ReLU) + bf16 convert for next layer's A operand
// ---------------------------------------------------------------------------
__global__ void post_kernel(const float* __restrict__ agg, const float* __restrict__ bias,
                            __bf16* __restrict__ hb, int do_relu) {
    int i = blockIdx.x * blockDim.x + threadIdx.x;
    if (i >= N_NODES * D) return;
    float v = agg[i] + bias[i & (D - 1)];
    if (do_relu) v = fmaxf(v, 0.f);
    hb[i] = (__bf16)v;
}

// ---------------------------------------------------------------------------
// 9) layer-3 epilogue: bias + pooled-sum per graph
// ---------------------------------------------------------------------------
__global__ void pool_kernel(const float* __restrict__ agg, const float* __restrict__ bias,
                            const int* __restrict__ batch, float* __restrict__ pooled) {
    int i = blockIdx.x * blockDim.x + threadIdx.x;
    if (i >= N_NODES * D) return;
    int node = i >> 7;
    int d    = i & (D - 1);
    float v = agg[i] + bias[d];
    atomicAdd(&pooled[(size_t)batch[node] * D + d], v);
}

// ---------------------------------------------------------------------------
// 10) final linear on pooled means: out[g,o] = (sum[g]/cnt[g]) . Wlin[:,o] + blin[o]
// ---------------------------------------------------------------------------
__global__ void final_kernel(const float* __restrict__ pooled, const float* __restrict__ cnt,
                             const float* __restrict__ Wlin, const float* __restrict__ blin,
                             float* __restrict__ out) {
    int tid = blockIdx.x * blockDim.x + threadIdx.x;
    if (tid >= N_GRAPHS * D_OUT) return;
    int g = tid / D_OUT, o = tid % D_OUT;
    float inv = 1.0f / fmaxf(cnt[g], 1.0f);
    float s = 0.f;
    for (int k = 0; k < D; ++k) s += pooled[g * D + k] * Wlin[k * D_OUT + o];
    out[tid] = s * inv + blin[o];
}

// ---------------------------------------------------------------------------
extern "C" void kernel_launch(void* const* d_in, const int* in_sizes, int n_in,
                              void* d_out, int out_size, void* d_ws, size_t ws_size,
                              hipStream_t stream) {
    const float* x      = (const float*)d_in[0];
    const int*   ei     = (const int*)  d_in[1];  // [2, E]
    const int*   batch  = (const int*)  d_in[2];
    const float* W1     = (const float*)d_in[3];
    const float* b1     = (const float*)d_in[4];
    const float* W2     = (const float*)d_in[5];
    const float* b2     = (const float*)d_in[6];
    const float* W3     = (const float*)d_in[7];
    const float* b3     = (const float*)d_in[8];
    const float* Wlin   = (const float*)d_in[9];
    const float* blin   = (const float*)d_in[10];
    float*       out    = (float*)d_out;

    const int* erow = ei;
    const int* ecol = ei + N_EDGES;

    // workspace layout (256-B aligned offsets)
    char* ws = (char*)d_ws;
    float*  dinv   = (float*)(ws + 0);                      // 50000*4      = 200000
    float*  counts = (float*)(ws + 200192);                 // 128*4
    float*  pooled = (float*)(ws + 200704);                 // 128*128*4    = 65536
    __bf16* wpack  = (__bf16*)(ws + 266240);                // 3*128*128*2  = 98304
    __bf16* hb     = (__bf16*)(ws + 364544);                // 50000*128*2  = 12800000
    float*  hw     = (float*)(ws + 13164544);               // 50000*128*4  = 25600000
    float*  agg    = (float*)(ws + 38764544);               // 50000*128*4  = 25600000
    // total: 64364544 bytes

    const int ND      = N_NODES * D;
    const int B       = 256;
    const int gN      = (N_NODES + B - 1) / B;
    const int gE      = (N_EDGES + B - 1) / B;
    const int gND     = (ND + B - 1) / B;
    const int gND4    = (ND / 4 + B - 1) / B;
    const int gGemm   = (N_NODES / 16 + 7) / 8;             // 8 waves/block
    const int gMsg    = (N_EDGES + N_NODES + 7) / 8;        // 1 wave/message

    init_kernel  <<<gN, B, 0, stream>>>(dinv, counts, pooled);
    degree_kernel<<<gE, B, 0, stream>>>(ecol, dinv);
    norm_kernel  <<<gN, B, 0, stream>>>(dinv, batch, counts);
    packw_kernel <<<3, 1024, 0, stream>>>(W1, W2, W3, wpack);
    cvt_kernel   <<<gND, B, 0, stream>>>(x, hb);

    // layer 1
    gemm_kernel   <<<gGemm, B, 0, stream>>>(hb, wpack, hw);
    zero_kernel   <<<gND4, B, 0, stream>>>((float4*)agg);
    scatter_kernel<<<gMsg, B, 0, stream>>>(hw, agg, erow, ecol, dinv);
    post_kernel   <<<gND, B, 0, stream>>>(agg, b1, hb, 1);

    // layer 2
    gemm_kernel   <<<gGemm, B, 0, stream>>>(hb, wpack + D * D, hw);
    zero_kernel   <<<gND4, B, 0, stream>>>((float4*)agg);
    scatter_kernel<<<gMsg, B, 0, stream>>>(hw, agg, erow, ecol, dinv);
    post_kernel   <<<gND, B, 0, stream>>>(agg, b2, hb, 1);

    // layer 3 (no ReLU) + global mean pool
    gemm_kernel   <<<gGemm, B, 0, stream>>>(hb, wpack + 2 * D * D, hw);
    zero_kernel   <<<gND4, B, 0, stream>>>((float4*)agg);
    scatter_kernel<<<gMsg, B, 0, stream>>>(hw, agg, erow, ecol, dinv);
    pool_kernel   <<<gND, B, 0, stream>>>(agg, b3, batch, pooled);

    final_kernel  <<<1, 384, 0, stream>>>(pooled, counts, Wlin, blin, out);
}
